// GuidedIterativeGraphUNet_63479616635387
// MI455X (gfx1250) — matrix operand order repaired and checked by reading, not compiled, not run on hardware
//
#include <hip/hip_runtime.h>
#include <hip/hip_bf16.h>

// ---------------- problem constants ----------------
constexpr int N_    = 20000;
constexpr int E_    = 640000;
constexpr int V_    = 8000;
constexpr int D_    = 256;
constexpr int T_    = 1024;
constexpr int MAXN_ = 20000;
constexpr int NH_   = 4;
constexpr int HD_   = 64;      // D_/NH_
constexpr int NE_   = E_ + N_; // edges + self loops
constexpr int FIN_  = 224;     // 128+32+32+32 input features

constexpr int LDSB_STRIDE = 36; // padded K-stride (bf16 elems) to spread LDS banks

typedef __attribute__((ext_vector_type(16))) __bf16 v16bf;
typedef __attribute__((ext_vector_type(8)))  float  v8f;

__device__ __forceinline__ float gelu_f(float x) {
    return 0.5f * x * (1.0f + erff(x * 0.70710678118654752f));
}

// ======================================================================
// bf16 WMMA GEMM:  C = act(alpha * (A @ B) + bias)
//   A: M x K  row-major (lda), bf16   -- M multiple of 32, K multiple of 32
//   B: K x Nc row-major (ldb), bf16   -- Nc multiple of 64
//   C: M x Nc f32 (ldc); optional bf16 mirror Cbf (ldcb)
// block = 128 threads = 4 waves. Workgroup tile = 128 rows x 64 cols:
//   wave w -> rows [row0 + 32w, +32) (2 A fragments, 8 accumulators),
//   all 64 cols. B chunk (32 x 64) is cooperatively loaded (coalesced
//   b128), stored TRANSPOSED into double-buffered LDS so each lane's B
//   fragment is read from two 16B runs of one LDS row; the next chunk's global loads
//   are issued before the current chunk's WMMAs to hide HBM latency.
// Fragment layouts per CDNA5 ISA.
// ======================================================================
__global__ __launch_bounds__(128)
void gemm_bf16_kernel(const __bf16* __restrict__ A, int lda,
                      const __bf16* __restrict__ B, int ldb,
                      const float*  __restrict__ bias,
                      float*        __restrict__ C, int ldc,
                      __bf16*       __restrict__ Cbf, int ldcb,
                      float alpha, int M, int Nc, int K, int act) {
    __shared__ __bf16 ldsB[2][64 * LDSB_STRIDE];   // [buf][n_local][k_local]

    const int tid  = threadIdx.x;
    const int wave = tid >> 5;
    const int lane = tid & 31;
    const int half = lane >> 4;   // 0/1
    const int l16  = lane & 15;

    const int col0  = blockIdx.x * 64;
    const int row0w = blockIdx.y * 128 + wave * 32;   // this wave's 32-row slice
    const bool rowok = (row0w < M);

    // cooperative-loader coordinates: thread t -> chunk row r (0..31), quarter qq (0..3)
    const int r  = tid >> 2;
    const int qq = tid & 3;

    v8f acc[8];
    #pragma unroll
    for (int j = 0; j < 8; ++j) acc[j] = (v8f){};

    __bf16 tmp[16];
    auto gload = [&](int kb) {
        const __bf16* gp = B + (size_t)(kb + r) * ldb + col0 + qq * 16;
        *(uint4*)(tmp)     = *(const uint4*)(gp);
        *(uint4*)(tmp + 8) = *(const uint4*)(gp + 8);
    };
    auto lstore = [&](int p) {
        #pragma unroll
        for (int i = 0; i < 16; ++i)
            ldsB[p][(qq * 16 + i) * LDSB_STRIDE + r] = tmp[i];
    };

    // prologue: stage chunk 0 into buffer 0
    gload(0);
    lstore(0);

    int p = 0;
    for (int kb = 0; kb < K; kb += 32) {
        __syncthreads();                        // buffer p is ready
        const bool more = (kb + 32 < K);
        if (more) gload(kb + 32);               // overlap next chunk's HBM loads

        if (rowok) {
            // ---- two A fragments (rows row0w..+15 and row0w+16..+31)
            const __bf16* Ap0 = A + (size_t)(row0w + l16) * lda + kb + 8 * half;
            const __bf16* Ap1 = Ap0 + (size_t)16 * lda;
            v16bf a0, a1;
            #pragma unroll
            for (int i = 0; i < 8; ++i) {
                a0[i] = Ap0[i]; a0[i + 8] = Ap0[16 + i];
                a1[i] = Ap1[i]; a1[i + 8] = Ap1[16 + i];
            }
            // ---- 4 B fragments from LDS: col = j*16 + l16, K = 8*half..+7 and 16+8*half..+7
            v16bf bf[4];
            #pragma unroll
            for (int j = 0; j < 4; ++j) {
                const __bf16* lp = &ldsB[p][(j * 16 + l16) * LDSB_STRIDE + 8 * half];
                #pragma unroll
                for (int e = 0; e < 16; ++e) bf[j][e] = lp[e < 8 ? e : e + 8];
            }
            // ---- 8 WMMAs: each B fragment feeds both row slices
            #pragma unroll
            for (int j = 0; j < 4; ++j) {
                acc[j]     = __builtin_amdgcn_wmma_f32_16x16x32_bf16(
                                 false, a0, false, bf[j], (short)0, acc[j], false, false);
                acc[4 + j] = __builtin_amdgcn_wmma_f32_16x16x32_bf16(
                                 false, a1, false, bf[j], (short)0, acc[4 + j], false, false);
            }
        }

        if (more) lstore(p ^ 1);                // fill the other buffer
        p ^= 1;
    }

    if (!rowok) return;

    // ---- C layout: VGPR rr -> M = rr + 8*half, N = l16 (per 16x16 tile)
    #pragma unroll
    for (int j = 0; j < 4; ++j) {
        const int n = col0 + j * 16 + l16;
        const float bb = bias ? bias[n] : 0.0f;
        #pragma unroll
        for (int rr = 0; rr < 8; ++rr) {
            {
                const int m = row0w + rr + 8 * half;
                float val = acc[j][rr] * alpha + bb;
                if (act == 1) val = gelu_f(val);
                C[(size_t)m * ldc + n] = val;
                if (Cbf) Cbf[(size_t)m * ldcb + n] = (__bf16)val;
            }
            {
                const int m = row0w + 16 + rr + 8 * half;
                float val = acc[4 + j][rr] * alpha + bb;
                if (act == 1) val = gelu_f(val);
                C[(size_t)m * ldc + n] = val;
                if (Cbf) Cbf[(size_t)m * ldcb + n] = (__bf16)val;
            }
        }
    }
}

// ---------------- utility kernels ----------------
__global__ void cast_bf16_kernel(const float* __restrict__ src,
                                 __bf16* __restrict__ dst, size_t n) {
    size_t i = (size_t)blockIdx.x * blockDim.x + threadIdx.x;
    if (i < n) dst[i] = (__bf16)src[i];
}

__global__ void fill_f32_kernel(float* __restrict__ p, float v, size_t n) {
    size_t i = (size_t)blockIdx.x * blockDim.x + threadIdx.x;
    if (i < n) p[i] = v;
}

// ---------------- embedding: build 224-dim bf16 node features ----------------
__global__ void embed_kernel(const float* __restrict__ x,
                             const float* __restrict__ tok_emb,
                             const float* __restrict__ prev_emb,
                             const float* __restrict__ iter_emb,
                             const float* __restrict__ pos_W,
                             const float* __restrict__ pos_b,
                             const int*   __restrict__ iter_ptr,
                             __bf16* __restrict__ feat) {
    const int nid = blockIdx.x;
    const int d   = threadIdx.x;
    if (d >= FIN_) return;
    const int it = iter_ptr[0];
    const int tok  = (int)x[nid * 5 + 0];
    const int prev = (int)x[nid * 5 + 1];
    const float px = x[nid * 5 + 2];
    const float py = x[nid * 5 + 3];
    float val;
    if (d < 128)       val = tok_emb[(size_t)tok * 128 + d];
    else if (d < 160)  val = prev_emb[(size_t)prev * 32 + (d - 128)];
    else if (d < 192) { int j = d - 160; val = px * pos_W[j] + py * pos_W[32 + j] + pos_b[j]; }
    else               val = iter_emb[(size_t)it * 32 + (d - 192)];
    feat[(size_t)nid * FIN_ + d] = (__bf16)val;
}

// ---------------- test embedding finish ----------------
__global__ void temb_finish_kernel(const float* __restrict__ t_lin,
                                   const int*   __restrict__ test_ids,
                                   const float* __restrict__ statuses,
                                   const float* __restrict__ id_emb,
                                   const float* __restrict__ st_W,
                                   const float* __restrict__ st_b,
                                   const float* __restrict__ tr_b,
                                   float* __restrict__ temb,
                                   __bf16* __restrict__ temb_bf) {
    const int t = blockIdx.x, d = threadIdx.x;
    int id = test_ids[t]; if (id < 0) id = 0;
    float val = t_lin[(size_t)t * D_ + d] + id_emb[(size_t)id * D_ + d]
              + statuses[t] * st_W[d] + st_b[d] + tr_b[d];
    val = gelu_f(val);
    temb[(size_t)t * D_ + d] = val;
    temb_bf[(size_t)t * D_ + d] = (__bf16)val;
}

// ---------------- GAT pieces ----------------
__global__ void gat_scores_kernel(const float* __restrict__ hw,
                                  const float* __restrict__ a_src,
                                  const float* __restrict__ a_dst,
                                  float* __restrict__ ssrc,
                                  float* __restrict__ sdst) {
    __shared__ float s1[D_], s2[D_];
    const int nid = blockIdx.x, d = threadIdx.x;
    const float h = hw[(size_t)nid * D_ + d];
    s1[d] = h * a_src[d];
    s2[d] = h * a_dst[d];
    __syncthreads();
    for (int st = D_ / 2; st > 0; st >>= 1) {
        if (d < st) { s1[d] += s1[d + st]; s2[d] += s2[d + st]; }
        __syncthreads();
    }
    if (d == 0) { ssrc[nid] = s1[0]; sdst[nid] = s2[0]; }
}

__device__ __forceinline__ void edge_endpoints(const int* ei, int e, int& s, int& dd) {
    if (e < E_) { s = ei[e]; dd = ei[E_ + e]; }
    else        { s = dd = e - E_; }     // self loops
}

__device__ __forceinline__ void atomicMaxFloat(float* addr, float val) {
    unsigned* ua = (unsigned*)addr;
    unsigned old = *ua;
    while (true) {
        float f = __uint_as_float(old);
        if (f >= val) break;
        unsigned assumed = old;
        old = atomicCAS(ua, assumed, __float_as_uint(val));
        if (old == assumed) break;
    }
}

__global__ void edge_logit_max_kernel(const int* __restrict__ ei,
                                      const float* __restrict__ ssrc,
                                      const float* __restrict__ sdst,
                                      float* __restrict__ lg,
                                      float* __restrict__ segmax) {
    const int e = blockIdx.x * blockDim.x + threadIdx.x;
    if (e >= NE_) return;
    int s, dd; edge_endpoints(ei, e, s, dd);
    float v = ssrc[s] + sdst[dd];
    v = (v > 0.0f) ? v : 0.2f * v;          // leaky_relu 0.2
    lg[e] = v;
    atomicMaxFloat(&segmax[dd], v);
}

__global__ void edge_exp_sum_kernel(const int* __restrict__ ei,
                                    const float* __restrict__ lg,
                                    const float* __restrict__ segmax,
                                    float* __restrict__ exbuf,
                                    float* __restrict__ segsum) {
    const int e = blockIdx.x * blockDim.x + threadIdx.x;
    if (e >= NE_) return;
    int s, dd; edge_endpoints(ei, e, s, dd);
    const float ex = expf(lg[e] - segmax[dd]);
    exbuf[e] = ex;
    atomicAdd(&segsum[dd], ex);
}

__global__ void edge_aggregate_kernel(const int* __restrict__ ei,
                                      const float* __restrict__ exbuf,
                                      const float* __restrict__ segsum,
                                      const float* __restrict__ hw,
                                      float* __restrict__ agg) {
    const int e = blockIdx.x;
    const int d = threadIdx.x;
    int s, dd; edge_endpoints(ei, e, s, dd);
    const float alpha = exbuf[e] / segsum[dd];
    atomicAdd(&agg[(size_t)dd * D_ + d], alpha * hw[(size_t)s * D_ + d]);
}

__global__ void gat_finish_kernel(const float* __restrict__ agg,
                                  const float* __restrict__ b,
                                  float* __restrict__ h,
                                  __bf16* __restrict__ hbf) {
    const int nid = blockIdx.x, d = threadIdx.x;
    const float v = gelu_f(agg[(size_t)nid * D_ + d] + b[d]);
    h[(size_t)nid * D_ + d] = v;
    hbf[(size_t)nid * D_ + d] = (__bf16)v;
}

// ---------------- cross-attention pieces ----------------
__global__ void transpose_head_kernel(const float* __restrict__ kf,
                                      __bf16* __restrict__ kT, int head) {
    const int gid = blockIdx.x * blockDim.x + threadIdx.x;
    if (gid >= HD_ * T_) return;
    const int kk = gid / T_;
    const int t  = gid % T_;
    kT[(size_t)kk * T_ + t] = (__bf16)kf[(size_t)t * D_ + head * HD_ + kk];
}

__global__ void attn_softmax_kernel(const float* __restrict__ s,
                                    const int* __restrict__ test_ids,
                                    __bf16* __restrict__ abf) {
    __shared__ float red[256];
    const int n = blockIdx.x, tid = threadIdx.x;
    float v[4];
    float mx = -1e30f;
    #pragma unroll
    for (int i = 0; i < 4; ++i) {
        const int t = tid + 256 * i;
        float x = s[(size_t)n * T_ + t];
        if (test_ids[t] == -1) x = -1e9f;     // mask
        v[i] = x;
        mx = fmaxf(mx, x);
    }
    red[tid] = mx; __syncthreads();
    for (int st = 128; st > 0; st >>= 1) {
        if (tid < st) red[tid] = fmaxf(red[tid], red[tid + st]);
        __syncthreads();
    }
    const float rowmax = red[0]; __syncthreads();
    float sum = 0.0f;
    #pragma unroll
    for (int i = 0; i < 4; ++i) { v[i] = expf(v[i] - rowmax); sum += v[i]; }
    red[tid] = sum; __syncthreads();
    for (int st = 128; st > 0; st >>= 1) {
        if (tid < st) red[tid] += red[tid + st];
        __syncthreads();
    }
    const float inv = 1.0f / red[0]; __syncthreads();
    #pragma unroll
    for (int i = 0; i < 4; ++i)
        abf[(size_t)n * T_ + tid + 256 * i] = (__bf16)(v[i] * inv);
}

__global__ void add_layernorm_kernel(const float* __restrict__ o,
                                     const float* __restrict__ g,
                                     const float* __restrict__ bln,
                                     float* __restrict__ h,
                                     __bf16* __restrict__ hbf) {
    __shared__ float red[D_];
    const int n = blockIdx.x, d = threadIdx.x;
    const float x = h[(size_t)n * D_ + d] + o[(size_t)n * D_ + d];
    red[d] = x; __syncthreads();
    for (int st = D_ / 2; st > 0; st >>= 1) { if (d < st) red[d] += red[d + st]; __syncthreads(); }
    const float mean = red[0] * (1.0f / D_); __syncthreads();
    const float c = x - mean;
    red[d] = c * c; __syncthreads();
    for (int st = D_ / 2; st > 0; st >>= 1) { if (d < st) red[d] += red[d + st]; __syncthreads(); }
    const float var = red[0] * (1.0f / D_); __syncthreads();
    const float y = c * rsqrtf(var + 1e-5f) * g[d] + bln[d];
    h[(size_t)n * D_ + d] = y;
    hbf[(size_t)n * D_ + d] = (__bf16)y;
}

__global__ void conf_kernel(const float* __restrict__ h,
                            const float* __restrict__ cW,
                            const float* __restrict__ cb,
                            float* __restrict__ out) {
    __shared__ float red[D_];
    const int n = blockIdx.x, d = threadIdx.x;
    red[d] = h[(size_t)n * D_ + d] * cW[d];
    __syncthreads();
    for (int st = D_ / 2; st > 0; st >>= 1) { if (d < st) red[d] += red[d + st]; __syncthreads(); }
    if (d == 0) out[n] = 1.0f / (1.0f + expf(-(red[0] + cb[0])));
}

// ======================================================================
// host orchestration
// ======================================================================
extern "C" void kernel_launch(void* const* d_in, const int* in_sizes, int n_in,
                              void* d_out, int out_size, void* d_ws, size_t ws_size,
                              hipStream_t stream) {
    (void)in_sizes; (void)n_in; (void)out_size; (void)ws_size;

    // ---- inputs (setup_inputs dict order, params flattened leaf-order) ----
    const float* x           = (const float*)d_in[0];
    const int*   edge_index  = (const int*)  d_in[1];
    const int*   test_ids    = (const int*)  d_in[2];
    const float* statuses    = (const float*)d_in[3];
    const float* traces      = (const float*)d_in[4];
    const int*   iter_ptr    = (const int*)  d_in[5];
    const float* tok_emb     = (const float*)d_in[6];
    const float* prev_emb    = (const float*)d_in[7];
    const float* iter_emb    = (const float*)d_in[8];
    const float* pos_W       = (const float*)d_in[9];
    const float* pos_b       = (const float*)d_in[10];
    const float* in_W        = (const float*)d_in[11];
    const float* in_b        = (const float*)d_in[12];
    const int GAT0 = 13, GAT1 = 17, GAT2 = 21, GAT3 = 25;  // enc0, bn0, bn1, dec0
    const int CA0 = 29, CA1 = 39;                          // ca0, ca_bn
    const float* id_emb      = (const float*)d_in[49];
    const float* st_W        = (const float*)d_in[50];
    const float* st_b        = (const float*)d_in[51];
    const float* tr_W        = (const float*)d_in[52];
    const float* tr_b        = (const float*)d_in[53];
    const float* pred_W      = (const float*)d_in[54];
    const float* pred_b      = (const float*)d_in[55];
    const float* conf_W      = (const float*)d_in[56];
    const float* conf_b      = (const float*)d_in[57];

    float* logits = (float*)d_out;                 // N x V
    float* confid = (float*)d_out + (size_t)N_ * V_;

    // ---- workspace carve ----
    char* base = (char*)d_ws;
    size_t off = 0;
    auto carve = [&](size_t bytes) -> void* {
        void* p = base + off;
        off = (off + bytes + 255) & ~(size_t)255;
        return p;
    };
    __bf16* featA   = (__bf16*)carve((size_t)N_ * FIN_ * 2);
    float*  h       = (float*) carve((size_t)N_ * D_ * 4);
    __bf16* h_bf    = (__bf16*)carve((size_t)N_ * D_ * 2);
    float*  hw      = (float*) carve((size_t)N_ * D_ * 4);
    float*  agg     = (float*) carve((size_t)N_ * D_ * 4);
    __bf16* wbuf    = (__bf16*)carve((size_t)MAXN_ * D_ * 2);  // largest weight (tr_W)
    __bf16* trc_bf  = (__bf16*)carve((size_t)T_ * MAXN_ * 2);
    float*  t_lin   = (float*) carve((size_t)T_ * D_ * 4);
    float*  temb    = (float*) carve((size_t)T_ * D_ * 4);
    __bf16* temb_bf = (__bf16*)carve((size_t)T_ * D_ * 2);
    float*  qf      = (float*) carve((size_t)N_ * D_ * 4);
    __bf16* q_bf    = (__bf16*)carve((size_t)N_ * D_ * 2);
    float*  kf      = (float*) carve((size_t)T_ * D_ * 4);
    float*  vf      = (float*) carve((size_t)T_ * D_ * 4);
    __bf16* v_bf    = (__bf16*)carve((size_t)T_ * D_ * 2);
    __bf16* kT_bf   = (__bf16*)carve((size_t)HD_ * T_ * 2);
    float*  sbuf    = (float*) carve((size_t)N_ * T_ * 4);
    __bf16* a_bf    = (__bf16*)carve((size_t)N_ * T_ * 2);
    float*  obuf    = (float*) carve((size_t)N_ * D_ * 4);
    __bf16* o_bf    = (__bf16*)carve((size_t)N_ * D_ * 2);
    float*  ssrc    = (float*) carve((size_t)N_ * 4);
    float*  sdst    = (float*) carve((size_t)N_ * 4);
    float*  segmax  = (float*) carve((size_t)N_ * 4);
    float*  segsum  = (float*) carve((size_t)N_ * 4);
    float*  lgbuf   = (float*) carve((size_t)NE_ * 4);
    float*  exbuf   = (float*) carve((size_t)NE_ * 4);

    // ---- helpers ----
    auto cast_bf16 = [&](const float* src, __bf16* dst, size_t n) {
        int blocks = (int)((n + 255) / 256);
        cast_bf16_kernel<<<blocks, 256, 0, stream>>>(src, dst, n);
    };
    auto fill = [&](float* p, float v, size_t n) {
        int blocks = (int)((n + 255) / 256);
        fill_f32_kernel<<<blocks, 256, 0, stream>>>(p, v, n);
    };
    auto gemm = [&](const __bf16* A, int lda, const __bf16* B, int ldb,
                    const float* bias, float* C, int ldc, __bf16* Cbf, int ldcb,
                    float alpha, int M, int Nc, int K, int act) {
        dim3 grid(Nc / 64, (M + 127) / 128);
        gemm_bf16_kernel<<<grid, 128, 0, stream>>>(A, lda, B, ldb, bias, C, ldc,
                                                   Cbf, ldcb, alpha, M, Nc, K, act);
    };

    // ================= 1. node embeddings + input MLP (gelu) =================
    embed_kernel<<<N_, 256, 0, stream>>>(x, tok_emb, prev_emb, iter_emb,
                                         pos_W, pos_b, iter_ptr, featA);
    cast_bf16(in_W, wbuf, (size_t)FIN_ * D_);
    gemm(featA, FIN_, wbuf, D_, in_b, h, D_, h_bf, D_, 1.0f, N_, D_, FIN_, 1);

    // ================= 2. test embeddings =================
    cast_bf16(traces, trc_bf, (size_t)T_ * MAXN_);
    cast_bf16(tr_W, wbuf, (size_t)MAXN_ * D_);
    gemm(trc_bf, MAXN_, wbuf, D_, nullptr, t_lin, D_, nullptr, 0,
         1.0f, T_, D_, MAXN_, 0);
    temb_finish_kernel<<<T_, D_, 0, stream>>>(t_lin, test_ids, statuses, id_emb,
                                              st_W, st_b, tr_b, temb, temb_bf);

    // ================= GAT layer runner =================
    auto run_gat = [&](int p) {
        const float* W     = (const float*)d_in[p + 0];
        const float* a_src = (const float*)d_in[p + 1];
        const float* a_dst = (const float*)d_in[p + 2];
        const float* b     = (const float*)d_in[p + 3];
        cast_bf16(W, wbuf, (size_t)D_ * D_);
        gemm(h_bf, D_, wbuf, D_, nullptr, hw, D_, nullptr, 0, 1.0f, N_, D_, D_, 0);
        gat_scores_kernel<<<N_, D_, 0, stream>>>(hw, a_src, a_dst, ssrc, sdst);
        fill(segmax, -1e30f, N_);
        fill(segsum, 0.0f, N_);
        fill(agg, 0.0f, (size_t)N_ * D_);
        const int eb = (NE_ + 255) / 256;
        edge_logit_max_kernel<<<eb, 256, 0, stream>>>(edge_index, ssrc, sdst, lgbuf, segmax);
        edge_exp_sum_kernel<<<eb, 256, 0, stream>>>(edge_index, lgbuf, segmax, exbuf, segsum);
        edge_aggregate_kernel<<<NE_, D_, 0, stream>>>(edge_index, exbuf, segsum, hw, agg);
        gat_finish_kernel<<<N_, D_, 0, stream>>>(agg, b, h, h_bf);   // gelu inside
    };

    // ================= cross-attention runner =================
    auto run_ca = [&](int p) {
        const float* Wq = (const float*)d_in[p + 0];
        const float* Wk = (const float*)d_in[p + 1];
        const float* Wv = (const float*)d_in[p + 2];
        const float* Wo = (const float*)d_in[p + 3];
        const float* bq = (const float*)d_in[p + 4];
        const float* bk = (const float*)d_in[p + 5];
        const float* bv = (const float*)d_in[p + 6];
        const float* bo = (const float*)d_in[p + 7];
        const float* lg = (const float*)d_in[p + 8];
        const float* lb = (const float*)d_in[p + 9];

        cast_bf16(Wq, wbuf, (size_t)D_ * D_);
        gemm(h_bf, D_, wbuf, D_, bq, qf, D_, q_bf, D_, 1.0f, N_, D_, D_, 0);
        cast_bf16(Wk, wbuf, (size_t)D_ * D_);
        gemm(temb_bf, D_, wbuf, D_, bk, kf, D_, nullptr, 0, 1.0f, T_, D_, D_, 0);
        cast_bf16(Wv, wbuf, (size_t)D_ * D_);
        gemm(temb_bf, D_, wbuf, D_, bv, vf, D_, v_bf, D_, 1.0f, T_, D_, D_, 0);

        for (int hd = 0; hd < NH_; ++hd) {
            transpose_head_kernel<<<(HD_ * T_ + 255) / 256, 256, 0, stream>>>(kf, kT_bf, hd);
            // scores: (N x 64) @ (64 x T), scaled by 1/sqrt(64)
            gemm(q_bf + hd * HD_, D_, kT_bf, T_, nullptr, sbuf, T_, nullptr, 0,
                 0.125f, N_, T_, HD_, 0);
            attn_softmax_kernel<<<N_, 256, 0, stream>>>(sbuf, test_ids, a_bf);
            // o_head: (N x T) @ (T x 64) into the head slice of obuf/o_bf
            gemm(a_bf, T_, v_bf + hd * HD_, D_, nullptr,
                 obuf + hd * HD_, D_, o_bf + hd * HD_, D_,
                 1.0f, N_, HD_, T_, 0);
        }
        cast_bf16(Wo, wbuf, (size_t)D_ * D_);
        gemm(o_bf, D_, wbuf, D_, bo, hw, D_, nullptr, 0, 1.0f, N_, D_, D_, 0);
        add_layernorm_kernel<<<N_, D_, 0, stream>>>(hw, lg, lb, h, h_bf);
    };

    // ================= pipeline =================
    run_gat(GAT0);   // enc0  (+gelu)
    run_ca (CA0);    // ca0
    run_gat(GAT1);   // bn0
    run_gat(GAT2);   // bn1
    run_ca (CA1);    // ca_bn
    run_gat(GAT3);   // dec0

    // ================= heads =================
    cast_bf16(pred_W, wbuf, (size_t)D_ * V_);
    gemm(h_bf, D_, wbuf, V_, pred_b, logits, V_, nullptr, 0, 1.0f, N_, V_, D_, 0);
    conf_kernel<<<N_, D_, 0, stream>>>(h, conf_W, conf_b, confid);
}
